// CrossAttention_59072980189646
// MI455X (gfx1250) — matrix operand order repaired and checked by reading, not compiled, not run on hardware
//
#include <hip/hip_runtime.h>

#define BATCH 4
#define SQ    1024
#define SKL   2048
#define DIM   1024
#define NH    16
#define DK    64

typedef __attribute__((ext_vector_type(16))) __bf16 v16bf;
typedef __attribute__((ext_vector_type(8)))  __bf16 v8bf;
typedef __attribute__((ext_vector_type(4)))  __bf16 v4bf;
typedef __attribute__((ext_vector_type(8)))  float  v8f;

__device__ __forceinline__ __bf16 cvt_bf16(float f) { return (__bf16)f; }

// A-matrix fragment (16x32 bf16, MxK): tile is row-major [M][K] with stride ld.
// lane L (0-15): row=L, K chunks {0..7, 16..23}; lane L+16: row=L, K {8..15, 24..31}.
__device__ __forceinline__ v16bf load_a_frag(const __bf16* tile, int ld, int lane) {
    const __bf16* p = tile + (size_t)(lane & 15) * ld + ((lane < 16) ? 0 : 8);
    v8bf lo = *(const v8bf*)(p);
    v8bf hi = *(const v8bf*)(p + 16);
    v16bf a;
#pragma unroll
    for (int i = 0; i < 8; ++i) { a[i] = lo[i]; a[i + 8] = hi[i]; }
    return a;
}

// B-matrix fragment (32x16 bf16, KxN): stored as [N][K] row-major with stride ld.
// lane L: col N = L&15, K chunks {0..7, 16..23}; lane L+16: K {8..15, 24..31}.
__device__ __forceinline__ v16bf load_b_frag(const __bf16* colbase, long ld, int lane) {
    const __bf16* p = colbase + (size_t)(lane & 15) * ld + ((lane < 16) ? 0 : 8);
    v8bf lo = *(const v8bf*)(p);
    v8bf hi = *(const v8bf*)(p + 16);
    v16bf r;
#pragma unroll
    for (int i = 0; i < 8; ++i) { r[i] = lo[i]; r[i + 8] = hi[i]; }
    return r;
}

// ---------------------------------------------------------------------------
// Generic NT GEMM: out[m][n] = sum_k X[m][k] * W[n][k] + bias[n]
//   IN_F32:  1 -> X is f32 (converted to bf16 on stage-in), 0 -> X is bf16.
//   OUT_MODE: 0 -> bf16 row-major, 1 -> bf16 transposed per batch
//             (out[b][n][sk], b=m/sklen), 2 -> f32 row-major.
// Tile: 64(M) x 64(N), 8 waves; each wave: one A-frag reused for two 16x16
// output tiles -> 2 WMMAs per K step per wave.
// ---------------------------------------------------------------------------
template <int IN_F32, int OUT_MODE>
__global__ void __launch_bounds__(256)
proj_gemm(const void* __restrict__ Xin, const float* __restrict__ W,
          const float* __restrict__ bias, void* __restrict__ outv,
          int N, int Kd, int sklen)
{
    __shared__ __align__(32) __bf16 As[64][32];
    __shared__ __align__(32) __bf16 Bs[64][32];

    const int tid = threadIdx.x;
    const int w = tid >> 5, lane = tid & 31;
    const int mt = w >> 1;              // 0..3: 16-row group
    const int ntp = (w & 1) * 2;        // 0 or 2: pair of 16-col groups
    const int col = lane & 15, rbase = (lane < 16) ? 0 : 8;
    const int m0 = blockIdx.x * 64;
    const int n0 = blockIdx.y * 64;
    const int lr = tid >> 3;            // 0..31
    const int lc = (tid & 7) * 4;       // 0..28

    v8f acc[2] = {{}, {}};
    for (int k0 = 0; k0 < Kd; k0 += 32) {
#pragma unroll
        for (int i = 0; i < 2; ++i) {
            int rr = lr + 32 * i;
            if (IN_F32) {
                const float* Xf = (const float*)Xin;
                float4 x = *(const float4*)(Xf + (size_t)(m0 + rr) * Kd + k0 + lc);
                As[rr][lc + 0] = cvt_bf16(x.x); As[rr][lc + 1] = cvt_bf16(x.y);
                As[rr][lc + 2] = cvt_bf16(x.z); As[rr][lc + 3] = cvt_bf16(x.w);
            } else {
                const __bf16* Xb = (const __bf16*)Xin;
                v4bf x = *(const v4bf*)(Xb + (size_t)(m0 + rr) * Kd + k0 + lc);
                As[rr][lc + 0] = x[0]; As[rr][lc + 1] = x[1];
                As[rr][lc + 2] = x[2]; As[rr][lc + 3] = x[3];
            }
            float4 ww = *(const float4*)(W + (size_t)(n0 + rr) * Kd + k0 + lc);
            Bs[rr][lc + 0] = cvt_bf16(ww.x); Bs[rr][lc + 1] = cvt_bf16(ww.y);
            Bs[rr][lc + 2] = cvt_bf16(ww.z); Bs[rr][lc + 3] = cvt_bf16(ww.w);
        }
        __syncthreads();
        v16bf a  = load_a_frag(&As[mt * 16][0], 32, lane);
        v16bf b0 = load_b_frag(&Bs[ntp * 16][0], 32, lane);
        v16bf b1 = load_b_frag(&Bs[(ntp + 1) * 16][0], 32, lane);
        acc[0] = __builtin_amdgcn_wmma_f32_16x16x32_bf16(false, a, false, b0,
                                                         (short)0, acc[0], false, false);
        acc[1] = __builtin_amdgcn_wmma_f32_16x16x32_bf16(false, a, false, b1,
                                                         (short)0, acc[1], false, false);
        __syncthreads();
    }

#pragma unroll
    for (int j = 0; j < 2; ++j) {
        int n = n0 + (ntp + j) * 16 + col;
        float bv = bias[n];
#pragma unroll
        for (int v = 0; v < 8; ++v) {
            int m = m0 + mt * 16 + rbase + v;
            float val = acc[j][v] + bv;
            if (OUT_MODE == 2) {
                ((float*)outv)[(size_t)m * N + n] = val;
            } else if (OUT_MODE == 0) {
                ((__bf16*)outv)[(size_t)m * N + n] = cvt_bf16(val);
            } else {
                int bb2 = m / sklen, sk = m % sklen;
                ((__bf16*)outv)[(size_t)bb2 * N * sklen + (size_t)n * sklen + sk] =
                    cvt_bf16(val);
            }
        }
    }
}

// ---------------------------------------------------------------------------
// Attention: one WG per (b, h, 16-row q block). 8 waves.
// Pass 1: each wave owns a 16-col k strip (stride 128), online (max,sum) in
//         registers via shfl reductions; combine across waves in LDS.
// Pass 2: recompute scores, write attn f32 to d_out, route P through LDS
//         (C-layout -> A-layout), accumulate ctx = P @ V via WMMA.
// ---------------------------------------------------------------------------
__global__ void __launch_bounds__(256)
attn_kernel(const __bf16* __restrict__ Qb, const __bf16* __restrict__ Kb,
            const __bf16* __restrict__ Vt, const int* __restrict__ maskp,
            float* __restrict__ attn_out, __bf16* __restrict__ ctxb)
{
    __shared__ __align__(32) __bf16 qs[16][64];
    __shared__ __align__(32) __bf16 pbuf[16][128];
    __shared__ float wmax[8][16];
    __shared__ float wsum[8][16];
    __shared__ float Mrow[16];
    __shared__ float Inv[16];
    __shared__ float ctxpart[2][16][64];

    const int tid = threadIdx.x;
    const int w = tid >> 5;
    const int lane = tid & 31;
    const int col = lane & 15;
    const int rbase = (lane < 16) ? 0 : 8;

    const int bx = blockIdx.x;
    const int qb = bx & 63;            // SQ/16 = 64 q-blocks
    const int h  = (bx >> 6) & 15;
    const int b  = bx >> 10;
    const int q0 = qb * 16;

    for (int i = tid; i < 16 * 64; i += 256) {
        int r = i >> 6, c = i & 63;
        qs[r][c] = Qb[(size_t)(b * SQ + q0 + r) * DIM + h * DK + c];
    }
    __syncthreads();

    const v16bf a0 = load_a_frag(&qs[0][0], 64, lane);
    const v16bf a1 = load_a_frag(&qs[0][32], 64, lane);
    const float scale = 0.125f;        // 1/sqrt(64)

    float mrun[8], srun[8];
#pragma unroll
    for (int v = 0; v < 8; ++v) { mrun[v] = -3.0e38f; srun[v] = 0.0f; }

    // ---- pass 1: row statistics ----
    for (int it = 0; it < SKL / 128; ++it) {
        int kbase = it * 128 + w * 16;
        const __bf16* kcol = Kb + (size_t)(b * SKL + kbase) * DIM + h * DK;
        if (it + 1 < SKL / 128)
            __builtin_prefetch(kcol + (size_t)128 * DIM, 0, 0);   // next K tile
        v16bf b0 = load_b_frag(kcol, DIM, lane);
        v16bf b1 = load_b_frag(kcol + 32, DIM, lane);
        v8f acc = {};
        acc = __builtin_amdgcn_wmma_f32_16x16x32_bf16(false, a0, false, b0,
                                                      (short)0, acc, false, false);
        acc = __builtin_amdgcn_wmma_f32_16x16x32_bf16(false, a1, false, b1,
                                                      (short)0, acc, false, false);
        int k = kbase + col;
#pragma unroll
        for (int v = 0; v < 8; ++v) {
            int q = q0 + rbase + v;
            float s = acc[v] * scale;
            if (maskp[(size_t)(b * SQ + q) * SKL + k] == 0) s = -1.0e9f;
            float tmax = s;
            tmax = fmaxf(tmax, __shfl_xor(tmax, 1));
            tmax = fmaxf(tmax, __shfl_xor(tmax, 2));
            tmax = fmaxf(tmax, __shfl_xor(tmax, 4));
            tmax = fmaxf(tmax, __shfl_xor(tmax, 8));
            float mnew = fmaxf(mrun[v], tmax);
            float e = __expf(s - mnew);
            e += __shfl_xor(e, 1);
            e += __shfl_xor(e, 2);
            e += __shfl_xor(e, 4);
            e += __shfl_xor(e, 8);
            srun[v] = srun[v] * __expf(mrun[v] - mnew) + e;
            mrun[v] = mnew;
        }
    }
    if (col == 0) {
#pragma unroll
        for (int v = 0; v < 8; ++v) {
            wmax[w][rbase + v] = mrun[v];
            wsum[w][rbase + v] = srun[v];
        }
    }
    __syncthreads();
    if (tid < 16) {
        float M = -3.0e38f;
        for (int ww = 0; ww < 8; ++ww) M = fmaxf(M, wmax[ww][tid]);
        float S = 0.0f;
        for (int ww = 0; ww < 8; ++ww) S += wsum[ww][tid] * __expf(wmax[ww][tid] - M);
        Mrow[tid] = M;
        Inv[tid] = 1.0f / S;
    }
    __syncthreads();

    // ---- pass 2: attn write + ctx = P @ V ----
    const int nt = w & 3;              // dk 16-col group
    const int khalf = w >> 2;          // which 64-k half of pbuf
    v8f cacc = {};
    float* attnBase = attn_out + ((size_t)(b * NH + h) * SQ + q0) * SKL;

    for (int it = 0; it < SKL / 128; ++it) {
        int kbase = it * 128 + w * 16;
        const __bf16* kcol = Kb + (size_t)(b * SKL + kbase) * DIM + h * DK;
        if (it + 1 < SKL / 128)
            __builtin_prefetch(kcol + (size_t)128 * DIM, 0, 0);   // next K tile
        v16bf b0 = load_b_frag(kcol, DIM, lane);
        v16bf b1 = load_b_frag(kcol + 32, DIM, lane);
        v8f acc = {};
        acc = __builtin_amdgcn_wmma_f32_16x16x32_bf16(false, a0, false, b0,
                                                      (short)0, acc, false, false);
        acc = __builtin_amdgcn_wmma_f32_16x16x32_bf16(false, a1, false, b1,
                                                      (short)0, acc, false, false);
        int k = kbase + col;
#pragma unroll
        for (int v = 0; v < 8; ++v) {
            int row = rbase + v;
            int q = q0 + row;
            float s = acc[v] * scale;
            if (maskp[(size_t)(b * SQ + q) * SKL + k] == 0) s = -1.0e9f;
            float p = __expf(s - Mrow[row]) * Inv[row];
            attnBase[(size_t)row * SKL + k] = p;
            pbuf[row][w * 16 + col] = cvt_bf16(p);
        }
        __syncthreads();
#pragma unroll
        for (int ks = 0; ks < 2; ++ks) {
            int koff = khalf * 64 + ks * 32;
            v16bf pa = load_a_frag(&pbuf[0][koff], 128, lane);
            const __bf16* vcol = Vt + ((size_t)b * DIM + h * DK + nt * 16) * SKL
                                 + (size_t)it * 128 + koff;
            if (it + 1 < SKL / 128)
                __builtin_prefetch(vcol + 128, 0, 0);             // next V tile
            v16bf vb = load_b_frag(vcol, SKL, lane);
            cacc = __builtin_amdgcn_wmma_f32_16x16x32_bf16(false, pa, false, vb,
                                                           (short)0, cacc, false, false);
        }
        __syncthreads();
    }

#pragma unroll
    for (int v = 0; v < 8; ++v)
        ctxpart[khalf][rbase + v][nt * 16 + col] = cacc[v];
    __syncthreads();
    for (int i = tid; i < 16 * 64; i += 256) {
        int r = i >> 6, c = i & 63;
        float val = ctxpart[0][r][c] + ctxpart[1][r][c];
        ctxb[(size_t)(b * SQ + q0 + r) * DIM + h * DK + c] = cvt_bf16(val);
    }
}

extern "C" void kernel_launch(void* const* d_in, const int* in_sizes, int n_in,
                              void* d_out, int out_size, void* d_ws, size_t ws_size,
                              hipStream_t stream)
{
    const float* dec = (const float*)d_in[0];
    const float* enc = (const float*)d_in[1];
    const int*   msk = (const int*)d_in[2];
    const float* W_Q = (const float*)d_in[3];
    const float* b_Q = (const float*)d_in[4];
    const float* W_K = (const float*)d_in[5];
    const float* b_K = (const float*)d_in[6];
    const float* W_V = (const float*)d_in[7];
    const float* b_V = (const float*)d_in[8];
    const float* W_O = (const float*)d_in[9];
    const float* b_O = (const float*)d_in[10];

    char* ws = (char*)d_ws;
    __bf16* Qb  = (__bf16*)(ws);                             //  8 MB: [4096][1024]
    __bf16* Kb  = (__bf16*)(ws + (size_t)8  * 1024 * 1024);  // 16 MB: [8192][1024]
    __bf16* Vt  = (__bf16*)(ws + (size_t)24 * 1024 * 1024);  // 16 MB: [4][1024][2048]
    __bf16* Ctx = (__bf16*)(ws + (size_t)40 * 1024 * 1024);  //  8 MB: [4096][1024]

    float* outp  = (float*)d_out;
    float* attnp = outp + (size_t)BATCH * SQ * DIM;

    dim3 blk(256);
    // Q = dec @ W_Q^T + b_Q  (row-major bf16)
    proj_gemm<1, 0><<<dim3(4096 / 64, 1024 / 64), blk, 0, stream>>>(
        dec, W_Q, b_Q, Qb, 1024, 1024, 1);
    // K = enc @ W_K^T + b_K  (row-major bf16)
    proj_gemm<1, 0><<<dim3(8192 / 64, 1024 / 64), blk, 0, stream>>>(
        enc, W_K, b_K, Kb, 1024, 1024, 1);
    // V = enc @ W_V^T + b_V, stored transposed per batch: Vt[b][d][sk]
    proj_gemm<1, 1><<<dim3(8192 / 64, 1024 / 64), blk, 0, stream>>>(
        enc, W_V, b_V, Vt, 1024, 1024, 2048);
    // attention: attn f32 -> d_out tail, ctx bf16 -> ws
    attn_kernel<<<dim3(4 * 16 * (SQ / 16)), blk, 0, stream>>>(
        Qb, Kb, Vt, msk, attnp, Ctx);
    // output = ctx @ W_O^T + b_O  (f32 -> d_out head)
    proj_gemm<0, 2><<<dim3(4096 / 64, 1024 / 64), blk, 0, stream>>>(
        Ctx, W_O, b_O, outp, 1024, 1024, 1);
}
